// SSMPhonotacticsModel_40441412059673
// MI455X (gfx1250) — compile-verified, matrix operand order checked by reading, not run on hardware
//
#include <hip/hip_runtime.h>
#include <math.h>

typedef __bf16 bf16_t;
typedef __bf16 v8bf  __attribute__((ext_vector_type(8)));
typedef __bf16 v16bf __attribute__((ext_vector_type(16)));
typedef float  v8f   __attribute__((ext_vector_type(8)));
typedef int    v4i   __attribute__((ext_vector_type(4)));

#define X_DIM 1024
#define S_DIM 128
#define T_LEN 128
#define BATCH 256

union FragB16 { v8bf h[2]; v16bf f; };

// ---- CDNA5 async global->LDS path (guarded; falls back to sync staging) ----
#if defined(__has_builtin)
#  if __has_builtin(__builtin_amdgcn_global_load_async_to_lds_b128) && \
      __has_builtin(__builtin_amdgcn_s_wait_asynccnt)
#    define USE_ASYNC_LDS 1
#  endif
#endif
#ifndef USE_ASYNC_LDS
#  define USE_ASYNC_LDS 0
#endif

#if USE_ASYNC_LDS
typedef __attribute__((address_space(1))) v4i as1_v4i;   // global b128 payload
typedef __attribute__((address_space(3))) v4i as3_v4i;   // LDS b128 payload
__device__ __forceinline__ void cp_async16(void* lds_dst, const void* gbl_src) {
  __builtin_amdgcn_global_load_async_to_lds_b128(
      (as1_v4i*)gbl_src, (as3_v4i*)lds_dst, 0, 0);
}
#endif

// ---------------------------------------------------------------------------
// Prep: A -> bf16 row-major [n][k] (GEMM "B" operand: (xA^T)[m,n]=sum_k x[m,k]A[n,k]);
// C -> bf16; B -> Bt bf16 (row s = B[:,s]); xs slot 0 = e0 per batch row.
// ---------------------------------------------------------------------------
__global__ __launch_bounds__(256)
void prep_kernel(const float* __restrict__ A, const float* __restrict__ B,
                 const float* __restrict__ C,
                 bf16_t* __restrict__ Abf, bf16_t* __restrict__ Cbf,
                 bf16_t* __restrict__ Btbf, bf16_t* __restrict__ xs0) {
  int i = blockIdx.x * blockDim.x + threadIdx.x;            // 0 .. 2^20-1
  if (i < X_DIM * X_DIM) Abf[i] = (bf16_t)A[i];
  if (i < S_DIM * X_DIM) {
    Cbf[i] = (bf16_t)C[i];
    int s = i >> 10, x = i & (X_DIM - 1);
    Btbf[i] = (bf16_t)B[x * S_DIM + s];
  }
  if (i < BATCH * X_DIM) {
    xs0[i] = (bf16_t)(((i & (X_DIM - 1)) == 0) ? 1.0f : 0.0f);
  }
}

// ---------------------------------------------------------------------------
// One recurrence step: xs_out[m][n] = sum_k xs_in[m][k]*A[n][k] + Bt[tok_m][n]
// Block tile 64(M) x 64(N), 4 waves in 2x2, wave tile 32x32
// -> four v_wmma_f32_16x16x32_bf16 per k-chunk of 32.
// LDS double-buffered, one barrier per k-chunk; async fills overlap compute.
// ---------------------------------------------------------------------------
__global__ __launch_bounds__(128)
void step_kernel(const bf16_t* __restrict__ xs_in, bf16_t* __restrict__ xs_out,
                 const bf16_t* __restrict__ W, const bf16_t* __restrict__ Btbf,
                 const int* __restrict__ tokens, int t) {
  __shared__ bf16_t tX[2][64 * 32];
  __shared__ bf16_t tW[2][64 * 32];

  const int tid  = threadIdx.x;           // 0..127
  const int lane = tid & 31;
  const int wave = tid >> 5;              // 0..3
  const int wm   = wave >> 1;             // 0..1 : M sub-slab of 32
  const int wn   = wave & 1;              // 0..1 : N sub-slab of 32
  const int bm0  = blockIdx.y * 64;
  const int bn0  = blockIdx.x * 64;

  const int fr = tid >> 1;                // 0..63  fill row
  const int fc = (tid & 1) * 16;          // 0 or 16 (elems); two 16B chunks each
  const int half = lane >> 4;             // 16-bit frag layout: hi lanes take k+8/k+24
  const int l15  = lane & 15;

  const bf16_t* gX = xs_in + (size_t)(bm0 + fr) * X_DIM + fc;
  const bf16_t* gW = W     + (size_t)(bn0 + fr) * X_DIM + fc;
  bf16_t* lX[2] = { &tX[0][fr * 32 + fc], &tX[1][fr * 32 + fc] };
  bf16_t* lW[2] = { &tW[0][fr * 32 + fc], &tW[1][fr * 32 + fc] };

  v8f acc00 = {}, acc01 = {}, acc10 = {}, acc11 = {};

  // Fill buffer `buf` with k-chunk starting at kk.
  auto fill = [&](int buf, int kk) {
#if USE_ASYNC_LDS
    cp_async16(lX[buf],     gX + kk);
    cp_async16(lX[buf] + 8, gX + kk + 8);
    cp_async16(lW[buf],     gW + kk);
    cp_async16(lW[buf] + 8, gW + kk + 8);
#else
    v8bf x0 = *(const v8bf*)(gX + kk);
    v8bf x1 = *(const v8bf*)(gX + kk + 8);
    v8bf w0 = *(const v8bf*)(gW + kk);
    v8bf w1 = *(const v8bf*)(gW + kk + 8);
    *(v8bf*)(lX[buf])     = x0;
    *(v8bf*)(lX[buf] + 8) = x1;
    *(v8bf*)(lW[buf])     = w0;
    *(v8bf*)(lW[buf] + 8) = w1;
#endif
    if (kk + 64 < X_DIM) {                // global_prefetch_b8 two chunks ahead
      __builtin_prefetch(gX + kk + 64, 0, 3);
      __builtin_prefetch(gW + kk + 64, 0, 3);
    }
  };

  fill(0, 0);

  for (int kc = 0; kc < X_DIM / 32; ++kc) {
    const int cur = kc & 1;
#if USE_ASYNC_LDS
    __builtin_amdgcn_s_wait_asynccnt(0);  // our async writes to buf `cur` landed
#endif
    __syncthreads();                      // everyone's writes visible; other buf free
    if (kc + 1 < X_DIM / 32) fill(cur ^ 1, (kc + 1) * 32);

    const bf16_t* xb = tX[cur];
    const bf16_t* wb = tW[cur];
    FragB16 a0, a1, b0, b1;
    const bf16_t* rA0 = xb + (wm * 32 + l15) * 32;
    a0.h[0] = *(const v8bf*)(rA0 + half * 8);
    a0.h[1] = *(const v8bf*)(rA0 + 16 + half * 8);
    const bf16_t* rA1 = xb + (wm * 32 + 16 + l15) * 32;
    a1.h[0] = *(const v8bf*)(rA1 + half * 8);
    a1.h[1] = *(const v8bf*)(rA1 + 16 + half * 8);
    const bf16_t* rB0 = wb + (wn * 32 + l15) * 32;
    b0.h[0] = *(const v8bf*)(rB0 + half * 8);
    b0.h[1] = *(const v8bf*)(rB0 + 16 + half * 8);
    const bf16_t* rB1 = wb + (wn * 32 + 16 + l15) * 32;
    b1.h[0] = *(const v8bf*)(rB1 + half * 8);
    b1.h[1] = *(const v8bf*)(rB1 + 16 + half * 8);

    acc00 = __builtin_amdgcn_wmma_f32_16x16x32_bf16(false, a0.f, false, b0.f,
                                                    (short)0, acc00, false, false);
    acc01 = __builtin_amdgcn_wmma_f32_16x16x32_bf16(false, a0.f, false, b1.f,
                                                    (short)0, acc01, false, false);
    acc10 = __builtin_amdgcn_wmma_f32_16x16x32_bf16(false, a1.f, false, b0.f,
                                                    (short)0, acc10, false, false);
    acc11 = __builtin_amdgcn_wmma_f32_16x16x32_bf16(false, a1.f, false, b1.f,
                                                    (short)0, acc11, false, false);
  }

  // Epilogue: D layout -> element (M = r + 8*half, N = l15); fuse B[:,tok] add.
  const int n0 = bn0 + wn * 32 + l15;           // tile j=0 column
  const int n1 = n0 + 16;                       // tile j=1 column
  const int mA = bm0 + wm * 32 + half * 8;      // tile i=0 row base
  const int mB = mA + 16;                       // tile i=1 row base
#pragma unroll
  for (int r = 0; r < 8; ++r) {
    int m0 = mA + r;
    int tok0 = tokens[m0 * T_LEN + t];
    const bf16_t* bt0 = Btbf + (size_t)tok0 * X_DIM;
    xs_out[(size_t)m0 * X_DIM + n0] = (bf16_t)(acc00[r] + (float)bt0[n0]);
    xs_out[(size_t)m0 * X_DIM + n1] = (bf16_t)(acc01[r] + (float)bt0[n1]);

    int m1 = mB + r;
    int tok1 = tokens[m1 * T_LEN + t];
    const bf16_t* bt1 = Btbf + (size_t)tok1 * X_DIM;
    xs_out[(size_t)m1 * X_DIM + n0] = (bf16_t)(acc10[r] + (float)bt1[n0]);
    xs_out[(size_t)m1 * X_DIM + n1] = (bf16_t)(acc11[r] + (float)bt1[n1]);
  }
}

// ---------------------------------------------------------------------------
// Logits + log-softmax + gather for all (t,b): rows r = t*256+b of xs,
// y[r][s] = sum_k xs[r][k] * C[s][k]. Block = 16 rows x 128 cols, wave w owns
// s-tile w; per-row softmax from LDS; ll written deterministically.
// ---------------------------------------------------------------------------
__global__ __launch_bounds__(256)
void logits_kernel(const bf16_t* __restrict__ xs, const bf16_t* __restrict__ Cbf,
                   const int* __restrict__ tokens, float* __restrict__ ll) {
  __shared__ float ylds[16 * 128];
  const int lane = threadIdx.x & 31;
  const int wave = threadIdx.x >> 5;
  const int half = lane >> 4;
  const int l15  = lane & 15;
  const size_t R0 = (size_t)blockIdx.x * 16;

  v8f acc = {};
  for (int k0 = 0; k0 < X_DIM; k0 += 32) {
    FragB16 a, b;
    const bf16_t* ra = xs + (R0 + l15) * X_DIM + k0;
    a.h[0] = *(const v8bf*)(ra + half * 8);
    a.h[1] = *(const v8bf*)(ra + 16 + half * 8);
    const bf16_t* rb = Cbf + (size_t)(wave * 16 + l15) * X_DIM + k0;
    b.h[0] = *(const v8bf*)(rb + half * 8);
    b.h[1] = *(const v8bf*)(rb + 16 + half * 8);
    acc = __builtin_amdgcn_wmma_f32_16x16x32_bf16(false, a.f, false, b.f,
                                                  (short)0, acc, false, false);
  }
#pragma unroll
  for (int r = 0; r < 8; ++r) {
    int m = half * 8 + r;
    ylds[m * 128 + wave * 16 + l15] = acc[r];
  }
  __syncthreads();

  if (threadIdx.x < 16) {
    const int m = threadIdx.x;
    const size_t rflat = R0 + m;
    const int tcur = (int)(rflat >> 8);      // / 256
    const int bcur = (int)(rflat & 255);
    const float* row = ylds + m * 128;
    float mx = row[0];
    for (int s = 1; s < S_DIM; ++s) mx = fmaxf(mx, row[s]);
    float sum = 0.0f;
    for (int s = 0; s < S_DIM; ++s) sum += __expf(row[s] - mx);
    const int tok = tokens[bcur * T_LEN + tcur];
    ll[rflat] = row[tok] - mx - __logf(sum);
  }
}

// Deterministic reduction over t (no float atomics -> bitwise reproducible).
__global__ __launch_bounds__(256)
void reduce_kernel(const float* __restrict__ ll, float* __restrict__ out) {
  const int b = threadIdx.x;
  float s = 0.0f;
  for (int t = 0; t < T_LEN; ++t) s += ll[(size_t)t * BATCH + b];
  out[b] = s;
}

// ---------------------------------------------------------------------------
extern "C" void kernel_launch(void* const* d_in, const int* in_sizes, int n_in,
                              void* d_out, int out_size, void* d_ws, size_t ws_size,
                              hipStream_t stream) {
  (void)in_sizes; (void)n_in; (void)out_size; (void)ws_size;
  const float* A      = (const float*)d_in[0];
  const float* B      = (const float*)d_in[1];
  const float* C      = (const float*)d_in[2];
  const int*   tokens = (const int*)d_in[3];
  float* out = (float*)d_out;

  char* ws = (char*)d_ws;
  constexpr size_t off_Abf  = 0;                          // 2 MiB
  constexpr size_t off_Cbf  = off_Abf  + (size_t)X_DIM * X_DIM * 2;
  constexpr size_t off_Btbf = off_Cbf  + (size_t)S_DIM * X_DIM * 2;
  constexpr size_t off_ll   = off_Btbf + (size_t)S_DIM * X_DIM * 2;
  constexpr size_t off_xs   = off_ll   + (size_t)T_LEN * BATCH * 4;

  bf16_t* Abf  = (bf16_t*)(ws + off_Abf);
  bf16_t* Cbf  = (bf16_t*)(ws + off_Cbf);
  bf16_t* Btbf = (bf16_t*)(ws + off_Btbf);
  float*  llb  = (float*)(ws + off_ll);
  bf16_t* xs   = (bf16_t*)(ws + off_xs);                  // 129 slots of 256x1024

  prep_kernel<<<4096, 256, 0, stream>>>(A, B, C, Abf, Cbf, Btbf, xs);

  constexpr size_t slot = (size_t)BATCH * X_DIM;
  for (int t = 0; t < T_LEN; ++t) {
    step_kernel<<<dim3(X_DIM / 64, BATCH / 64), 128, 0, stream>>>(
        xs + (size_t)t * slot, xs + (size_t)(t + 1) * slot, Abf, Btbf, tokens, t);
  }

  logits_kernel<<<(T_LEN * BATCH) / 16, 256, 0, stream>>>(xs, Cbf, tokens, llb);
  reduce_kernel<<<1, 256, 0, stream>>>(llb, out);
}
